// MultiAttentionLayer_41549513621919
// MI455X (gfx1250) — matrix operand order repaired and checked by reading, not compiled, not run on hardware
//
#include <hip/hip_runtime.h>

typedef __attribute__((ext_vector_type(16))) _Float16 v16h;
typedef __attribute__((ext_vector_type(8)))  float    v8f;

#define WMMA_F16(a,b,c) \
  __builtin_amdgcn_wmma_f32_16x16x32_f16(false,(a),false,(b),(short)0,(c),false,false)

// ---- CDNA5 wave32 WMMA fragment maps (CDNA5 ISA §7.12.2) ----
// A (16x32 f16): m = lane&15 ; element e (0..15):
//   e<8 : K = (lane<16?0:8)+e     (VGPR0..3)
//   e>=8: K = 16+(lane<16?0:8)+(e-8)  (VGPR4..7)
__device__ __forceinline__ int kmapA(int lane, int e) {
  int base = (lane < 16) ? 0 : 8;
  return (e < 8) ? (base + e) : (16 + base + (e - 8));
}
// B (32x16 f16): n = lane&15 ; K same as A: e<8 : (lane<16?0:8)+e ; e>=8: 16+(lane<16?0:8)+(e-8)
__device__ __forceinline__ int kmapB(int lane, int e) {
  int base = (lane < 16) ? 0 : 8;
  return (e < 8) ? (base + e) : (16 + base + (e - 8));
}
// C/D (16x16 f32): reg r -> row (lane<16 ? r : r+8), col lane&15

// =====================================================================
// Weight pre-pack: f32 row-major (Kd x Nd) -> f16 fragments, lane-major.
// Fragment (nt,kc): dst[((nt*(Kd/32)+kc)*32 + lane)*16 + e]
//   = (f16) src[(kc*32 + kmapB(lane,e))*Nd + nt*16 + (lane&15)]
// Each lane's 16 halves are 32B contiguous -> one v16h load at use time.
// =====================================================================
__global__ void pack_b_kernel(const float* __restrict__ src,
                              _Float16* __restrict__ dst,
                              int Kd, int Nd, int nmat) {
  int tid = blockIdx.x * blockDim.x + threadIdx.x;
  int fragsPerMat = (Nd / 16) * (Kd / 32);
  int total = nmat * fragsPerMat * 32;
  if (tid >= total) return;
  int lane = tid & 31;
  int frag = (tid >> 5) % fragsPerMat;
  int mat  = (tid >> 5) / fragsPerMat;
  int numKc = Kd / 32;
  int nt = frag / numKc, kc = frag % numKc;
  const float* s = src + (size_t)mat * Kd * Nd;
  _Float16* d = dst + (size_t)mat * Kd * Nd + ((size_t)frag * 32 + lane) * 16;
  int n = nt * 16 + (lane & 15);
#pragma unroll
  for (int e = 0; e < 16; ++e) {
    int k = kc * 32 + kmapB(lane, e);
    d[e] = (_Float16)s[(size_t)k * Nd + n];
  }
}

// =====================================================================
// KNN: top-16 smallest squared distances (branchless sorted insertion).
// 4 blocks per batch; x[b] (12KB) staged in LDS.
// =====================================================================
__global__ __launch_bounds__(256) void knn_kernel(const float* __restrict__ x,
                                                  int* __restrict__ nn_idx) {
  __shared__ float xs[1024 * 3];
  int b = blockIdx.x >> 2;
  int chunk = blockIdx.x & 3;
  const float* xb = x + (size_t)b * 1024 * 3;
  for (int i = threadIdx.x; i < 1024 * 3; i += blockDim.x) xs[i] = xb[i];
  __syncthreads();
  int n = chunk * 256 + threadIdx.x;
  float x0 = xs[n * 3 + 0], x1 = xs[n * 3 + 1], x2 = xs[n * 3 + 2];
  float bd[16]; int bi[16];
#pragma unroll
  for (int i = 0; i < 16; ++i) { bd[i] = 3.4e38f; bi[i] = 0x7fffffff; }
  for (int m = 0; m < 1024; ++m) {
    float d0 = xs[m * 3 + 0] - x0;
    float d1 = xs[m * 3 + 1] - x1;
    float d2 = xs[m * 3 + 2] - x2;
    float d = d0 * d0 + d1 * d1 + d2 * d2;
    if (d < bd[15] || (d == bd[15] && m < bi[15])) {
      float cd = d; int ci = m;
#pragma unroll
      for (int i = 0; i < 16; ++i) {
        bool ins = (cd < bd[i]) || (cd == bd[i] && ci < bi[i]);
        float td = ins ? bd[i] : cd; int ti = ins ? bi[i] : ci;
        bd[i] = ins ? cd : bd[i];   bi[i] = ins ? ci : bi[i];
        cd = td; ci = ti;
      }
    }
  }
  int* outp = nn_idx + ((size_t)b * 1024 + n) * 16;
#pragma unroll
  for (int i = 0; i < 16; ++i) outp[i] = bi[i];
}

// ===================== xl = x @ w_in (K=3, VALU) ======================
__global__ void xl_kernel(const float* __restrict__ x,
                          const float* __restrict__ w_in,
                          float* __restrict__ xl) {
  int t = blockIdx.x * blockDim.x + threadIdx.x;  // 4096*128 threads
  int r = t >> 7, u = t & 127;
  const float* xr = x + (size_t)r * 3;
  xl[t] = xr[0] * w_in[u] + xr[1] * w_in[128 + u] + xr[2] * w_in[256 + u];
}

// ====== q/k/v = xl @ {wq,wk,wv} : one wave per 16x16 tile, WMMA ======
__global__ __launch_bounds__(128) void qkv_kernel(
    const float* __restrict__ xl,
    const _Float16* __restrict__ pwq, const _Float16* __restrict__ pwk,
    const _Float16* __restrict__ pwv,
    float* __restrict__ q, float* __restrict__ k, float* __restrict__ v) {
  int wid  = (blockIdx.x * blockDim.x + threadIdx.x) >> 5;
  int lane = threadIdx.x & 31;
  int which = wid / (256 * 32);
  int rem   = wid % (256 * 32);
  int mt = rem >> 5, nt = rem & 31;
  const _Float16* pw = (which == 0) ? pwq : (which == 1) ? pwk : pwv;
  float* dst = (which == 0) ? q : (which == 1) ? k : v;
  int m = lane & 15;
  v8f acc = {};
#pragma unroll
  for (int kc = 0; kc < 4; ++kc) {            // Kd=128
    v16h a;
    const float* arow = xl + (size_t)(mt * 16 + m) * 128 + kc * 32;
#pragma unroll
    for (int e = 0; e < 16; ++e) a[e] = (_Float16)arow[kmapA(lane, e)];
    v16h bf = *(const v16h*)(pw + ((size_t)(nt * 4 + kc) * 32 + lane) * 16);
    acc = WMMA_F16(a, bf, acc);
  }
  int nc = nt * 16 + (lane & 15);
#pragma unroll
  for (int r = 0; r < 8; ++r) {
    int mm = mt * 16 + ((lane < 16) ? r : r + 8);
    dst[(size_t)mm * 512 + nc] = acc[r];
  }
}

// =====================================================================
// Fused attention: one wave per point (b,n). K=16 neighbors == one WMMA
// M-tile. Stages: edge-MLP L1 (VALU) -> pos via WMMA -> per head:
// att_in -> relu(@a0) -> @a1 -> softmax over K -> weighted val sum.
// =====================================================================
struct WaveSmem {
  float posb[16 * 128];   // pos (K x U), shared across heads
  float attin[16 * 128];  // q - k_gather + pos
  float attb[16 * 128];   // att logits after a1
  float midb[16 * 64];    // pos1 / relu(a0) staging
  float qrow[128];
  int   idxs[16];
};

__global__ __launch_bounds__(64) void attn_kernel(
    const float* __restrict__ x, const float* __restrict__ pos_w0,
    const _Float16* __restrict__ pw1, const _Float16* __restrict__ pa0,
    const _Float16* __restrict__ pa1, const int* __restrict__ nn_idx,
    const float* __restrict__ q, const float* __restrict__ kmat,
    const float* __restrict__ vmat, float* __restrict__ mh) {
  __shared__ WaveSmem sm[2];
  int lane = threadIdx.x & 31;
  int wslot = threadIdx.x >> 5;
  WaveSmem& S = sm[wslot];
  int p = blockIdx.x * 2 + wslot;          // 0..4095
  int b = p >> 10, n = p & 1023;

  const int* idxp = nn_idx + (size_t)p * 16;
  if (lane < 16) S.idxs[lane] = idxp[lane];
  __syncthreads();

  const float* xb = x + (size_t)b * 1024 * 3;
  float cx = xb[n * 3 + 0], cy = xb[n * 3 + 1], cz = xb[n * 3 + 2];

  // pos1 = relu(edges @ pos_w0)  (16 x 64), lane covers 2 cols/row
  for (int kk = 0; kk < 16; ++kk) {
    int j = S.idxs[kk];
    float e0 = xb[j * 3 + 0] - cx, e1 = xb[j * 3 + 1] - cy, e2 = xb[j * 3 + 2] - cz;
#pragma unroll
    for (int t = 0; t < 2; ++t) {
      int c = lane + t * 32;
      float s = e0 * pos_w0[c] + e1 * pos_w0[64 + c] + e2 * pos_w0[128 + c];
      S.midb[kk * 64 + c] = fmaxf(s, 0.f);
    }
  }
  __syncthreads();

  // pos = pos1 @ pos_w1  (Kd=64 -> 2 kc, Nd=128 -> 8 nt)
#pragma unroll
  for (int nt = 0; nt < 8; ++nt) {
    v8f acc = {};
#pragma unroll
    for (int kc = 0; kc < 2; ++kc) {
      v16h a; int m = lane & 15;
#pragma unroll
      for (int e = 0; e < 16; ++e)
        a[e] = (_Float16)S.midb[m * 64 + kc * 32 + kmapA(lane, e)];
      v16h bf = *(const v16h*)(pw1 + ((size_t)(nt * 2 + kc) * 32 + lane) * 16);
      acc = WMMA_F16(a, bf, acc);
    }
    int nc = nt * 16 + (lane & 15);
#pragma unroll
    for (int r = 0; r < 8; ++r)
      S.posb[((lane < 16) ? r : r + 8) * 128 + nc] = acc[r];
  }
  __syncthreads();

  size_t rowbase = (size_t)p * 512;
  for (int h = 0; h < 4; ++h) {
    const float* qp = q + rowbase + h * 128;
#pragma unroll
    for (int t = 0; t < 4; ++t) S.qrow[lane + t * 32] = qp[lane + t * 32];
    __syncthreads();

    // att_in = q - gather(k) + pos
    for (int kk = 0; kk < 16; ++kk) {
      size_t kb = (size_t)(b * 1024 + S.idxs[kk]) * 512 + h * 128;
#pragma unroll
      for (int t = 0; t < 4; ++t) {
        int u = lane + t * 32;
        S.attin[kk * 128 + u] = S.qrow[u] - kmat[kb + u] + S.posb[kk * 128 + u];
      }
    }
    __syncthreads();

    // mid = relu(att_in @ a0[h])  (Kd=128 -> 4 kc, Nd=64 -> 4 nt)
    const _Float16* pa0h = pa0 + (size_t)h * 128 * 64;
#pragma unroll
    for (int nt = 0; nt < 4; ++nt) {
      v8f acc = {};
#pragma unroll
      for (int kc = 0; kc < 4; ++kc) {
        v16h a; int m = lane & 15;
#pragma unroll
        for (int e = 0; e < 16; ++e)
          a[e] = (_Float16)S.attin[m * 128 + kc * 32 + kmapA(lane, e)];
        v16h bf = *(const v16h*)(pa0h + ((size_t)(nt * 4 + kc) * 32 + lane) * 16);
        acc = WMMA_F16(a, bf, acc);
      }
      int nc = nt * 16 + (lane & 15);
#pragma unroll
      for (int r = 0; r < 8; ++r)
        S.midb[((lane < 16) ? r : r + 8) * 64 + nc] = fmaxf(acc[r], 0.f);
    }
    __syncthreads();

    // att = mid @ a1[h]  (Kd=64 -> 2 kc, Nd=128 -> 8 nt)
    const _Float16* pa1h = pa1 + (size_t)h * 64 * 128;
#pragma unroll
    for (int nt = 0; nt < 8; ++nt) {
      v8f acc = {};
#pragma unroll
      for (int kc = 0; kc < 2; ++kc) {
        v16h a; int m = lane & 15;
#pragma unroll
        for (int e = 0; e < 16; ++e)
          a[e] = (_Float16)S.midb[m * 64 + kc * 32 + kmapA(lane, e)];
        v16h bf = *(const v16h*)(pa1h + ((size_t)(nt * 2 + kc) * 32 + lane) * 16);
        acc = WMMA_F16(a, bf, acc);
      }
      int nc = nt * 16 + (lane & 15);
#pragma unroll
      for (int r = 0; r < 8; ++r)
        S.attb[((lane < 16) ? r : r + 8) * 128 + nc] = acc[r];
    }
    __syncthreads();

    // softmax over K (rows) per column + weighted sum of (v_gather + pos)
    float* mout = mh + rowbase + h * 128;
    for (int t = 0; t < 4; ++t) {
      int u = lane + t * 32;
      float mx = -3.4e38f;
#pragma unroll
      for (int kk = 0; kk < 16; ++kk) mx = fmaxf(mx, S.attb[kk * 128 + u]);
      float sum = 0.f;
#pragma unroll
      for (int kk = 0; kk < 16; ++kk) sum += __expf(S.attb[kk * 128 + u] - mx);
      float inv = 1.f / sum;
      float acc = 0.f;
      for (int kk = 0; kk < 16; ++kk) {
        size_t vb = (size_t)(b * 1024 + S.idxs[kk]) * 512 + h * 128 + u;
        float w = __expf(S.attb[kk * 128 + u] - mx) * inv;
        acc += (vmat[vb] + S.posb[kk * 128 + u]) * w;
      }
      mout[u] = acc;
    }
    __syncthreads();
  }
}

// ============ out = multi_heads @ w_out (4096x512 @ 512x128) ==========
__global__ __launch_bounds__(128) void out_kernel(
    const float* __restrict__ mhm, const _Float16* __restrict__ pwo,
    float* __restrict__ out) {
  int wid = (blockIdx.x * blockDim.x + threadIdx.x) >> 5;
  int lane = threadIdx.x & 31;
  int mt = wid >> 3, nt = wid & 7;
  int m = lane & 15;
  v8f acc = {};
#pragma unroll
  for (int kc = 0; kc < 16; ++kc) {           // Kd=512
    v16h a;
    const float* arow = mhm + (size_t)(mt * 16 + m) * 512 + kc * 32;
#pragma unroll
    for (int e = 0; e < 16; ++e) a[e] = (_Float16)arow[kmapA(lane, e)];
    v16h bf = *(const v16h*)(pwo + ((size_t)(nt * 16 + kc) * 32 + lane) * 16);
    acc = WMMA_F16(a, bf, acc);
  }
  int nc = nt * 16 + (lane & 15);
#pragma unroll
  for (int r = 0; r < 8; ++r) {
    int mm = mt * 16 + ((lane < 16) ? r : r + 8);
    out[(size_t)mm * 128 + nc] = acc[r];
  }
}

// =====================================================================
extern "C" void kernel_launch(void* const* d_in, const int* in_sizes, int n_in,
                              void* d_out, int out_size, void* d_ws, size_t ws_size,
                              hipStream_t stream) {
  (void)in_sizes; (void)n_in; (void)out_size; (void)ws_size;
  const float* x      = (const float*)d_in[0];
  const float* pos_w0 = (const float*)d_in[1];
  const float* pos_w1 = (const float*)d_in[2];
  const float* w_in   = (const float*)d_in[3];
  const float* wq     = (const float*)d_in[4];
  const float* wk     = (const float*)d_in[5];
  const float* wv     = (const float*)d_in[6];
  const float* a0     = (const float*)d_in[7];
  const float* a1     = (const float*)d_in[8];
  const float* w_out  = (const float*)d_in[9];
  float* out = (float*)d_out;

  char* ws = (char*)d_ws;
  size_t off = 0;
  auto take = [&](size_t bytes) -> char* {
    off = (off + 255) & ~(size_t)255;
    char* p = ws + off;
    off += bytes;
    return p;
  };
  int*      d_idx = (int*)take((size_t)4096 * 16 * 4);
  float*    d_xl  = (float*)take((size_t)4096 * 128 * 4);
  float*    d_q   = (float*)take((size_t)4096 * 512 * 4);
  float*    d_k   = (float*)take((size_t)4096 * 512 * 4);
  float*    d_v   = (float*)take((size_t)4096 * 512 * 4);
  float*    d_mh  = (float*)take((size_t)4096 * 512 * 4);
  _Float16* p_wq  = (_Float16*)take((size_t)128 * 512 * 2);
  _Float16* p_wk  = (_Float16*)take((size_t)128 * 512 * 2);
  _Float16* p_wv  = (_Float16*)take((size_t)128 * 512 * 2);
  _Float16* p_w1  = (_Float16*)take((size_t)64 * 128 * 2);
  _Float16* p_a0  = (_Float16*)take((size_t)4 * 128 * 64 * 2);
  _Float16* p_a1  = (_Float16*)take((size_t)4 * 64 * 128 * 2);
  _Float16* p_wo  = (_Float16*)take((size_t)512 * 128 * 2);

  auto packGrid = [](int Kd, int Nd, int nmat) {
    int threads = nmat * Kd * Nd / 16;
    return (threads + 255) / 256;
  };
  pack_b_kernel<<<packGrid(128, 512, 1), 256, 0, stream>>>(wq, p_wq, 128, 512, 1);
  pack_b_kernel<<<packGrid(128, 512, 1), 256, 0, stream>>>(wk, p_wk, 128, 512, 1);
  pack_b_kernel<<<packGrid(128, 512, 1), 256, 0, stream>>>(wv, p_wv, 128, 512, 1);
  pack_b_kernel<<<packGrid(64, 128, 1), 256, 0, stream>>>(pos_w1, p_w1, 64, 128, 1);
  pack_b_kernel<<<packGrid(128, 64, 4), 256, 0, stream>>>(a0, p_a0, 128, 64, 4);
  pack_b_kernel<<<packGrid(64, 128, 4), 256, 0, stream>>>(a1, p_a1, 64, 128, 4);
  pack_b_kernel<<<packGrid(512, 128, 1), 256, 0, stream>>>(w_out, p_wo, 512, 128, 1);

  knn_kernel<<<16, 256, 0, stream>>>(x, d_idx);
  xl_kernel<<<(4096 * 128) / 256, 256, 0, stream>>>(x, w_in, d_xl);
  // 3 outputs * 256 mtiles * 32 ntiles = 24576 waves; 4 waves/block
  qkv_kernel<<<24576 / 4, 128, 0, stream>>>(d_xl, p_wq, p_wk, p_wv, d_q, d_k, d_v);
  // one wave per point, 2 waves/block
  attn_kernel<<<2048, 64, 0, stream>>>(x, pos_w0, p_w1, p_a0, p_a1, d_idx,
                                       d_q, d_k, d_v, d_mh);
  // 256 mtiles * 8 ntiles = 2048 waves; 4 waves/block
  out_kernel<<<2048 / 4, 128, 0, stream>>>(d_mh, p_wo, out);
}
